// SelfAttention_17351667876226
// MI455X (gfx1250) — matrix operand order repaired and checked by reading, not compiled, not run on hardware
//
#include <hip/hip_runtime.h>
#include <math.h>

typedef _Float16 half_t;
typedef __attribute__((ext_vector_type(16))) _Float16 v16h;
typedef __attribute__((ext_vector_type(8)))  float    v8f;

#define B_ 4
#define N_ 4096
#define C_ 128
#define D_ 16
#define NW 4                                  // split-K waves per query tile

// ---------------------------------------------------------------------------
// Kernel 1: 1x1-conv projections.  Q (pre-scaled by 1/ln2) and K -> f16
// [B,N,16]; V -> f16 TRANSPOSED [B,C,N] for contiguous WMMA A-operand loads.
// ---------------------------------------------------------------------------
__global__ __launch_bounds__(128)
void proj_kernel(const float* __restrict__ x,
                 const float* __restrict__ wq, const float* __restrict__ bq,
                 const float* __restrict__ wk, const float* __restrict__ bk,
                 const float* __restrict__ wv, const float* __restrict__ bv,
                 half_t* __restrict__ Qh, half_t* __restrict__ Kh,
                 half_t* __restrict__ Vt)
{
    __shared__ float xs[16][C_];
    const int blk = blockIdx.x;              // 0 .. B*N/16-1
    const int b   = blk / (N_ / 16);
    const int n0  = (blk % (N_ / 16)) * 16;
    const int t   = threadIdx.x;             // 0..127

    #pragma unroll
    for (int i = 0; i < 16; ++i)
        xs[i][t] = x[((size_t)(b * N_ + n0 + i)) * C_ + t];
    __syncthreads();

    float acc[16];
    #pragma unroll
    for (int i = 0; i < 16; ++i) acc[i] = bv[t];
    for (int c = 0; c < C_; ++c) {
        const float w = wv[c * C_ + t];
        #pragma unroll
        for (int i = 0; i < 16; ++i) acc[i] = fmaf(xs[i][c], w, acc[i]);
    }
    half_t* vrow = Vt + ((size_t)(b * C_ + t)) * N_ + n0;   // transposed store
    #pragma unroll
    for (int i = 0; i < 16; ++i) vrow[i] = (half_t)acc[i];

    if (t < 32) {
        const int     ch = t & 15;
        const float*  w  = (t < 16) ? wq : wk;
        const float*  bb = (t < 16) ? bq : bk;
        const float   sc = (t < 16) ? 1.44269504088896340736f : 1.0f;
        float a[16];
        #pragma unroll
        for (int i = 0; i < 16; ++i) a[i] = bb[ch];
        for (int c = 0; c < C_; ++c) {
            const float ww = w[c * D_ + ch];
            #pragma unroll
            for (int i = 0; i < 16; ++i) a[i] = fmaf(xs[i][c], ww, a[i]);
        }
        half_t* dst = (t < 16) ? Qh : Kh;
        #pragma unroll
        for (int i = 0; i < 16; ++i)
            dst[(size_t)(b * N_ + n0 + i) * D_ + ch] = (half_t)(a[i] * sc);
    }
}

// ---------------------------------------------------------------------------
// Kernel 2: split-K flash attention.  4 waves per 16-query tile; wave w
// processes key strips kt = w, w+4, ...  Per strip (32 keys):
//   S^T  = Ktile x Q^T  (2x wmma; key->A-row remap, in-lane softmax)
//   O^T += V^T x P^T    (8x wmma, FULL K=32; zero cross-lane exchange)
// Partials (m, l, O^T) merged once through LDS in the exp2 domain.
// ---------------------------------------------------------------------------
__global__ __launch_bounds__(32 * NW)
void attn_kernel(const half_t* __restrict__ Qh, const half_t* __restrict__ Kh,
                 const half_t* __restrict__ Vt,
                 const float* __restrict__ x, const float* __restrict__ gamma,
                 float* __restrict__ out)
{
    __shared__ float Osts[NW][16][C_ + 4];   // per-wave scaled partial O^T
    __shared__ float Ml[NW][16], Ll[NW][16], Lst[16];

    const int blk  = blockIdx.x;             // 0 .. B*N/16-1
    const int b    = blk / (N_ / 16);
    const int row0 = (blk % (N_ / 16)) * 16;
    const int tid  = threadIdx.x;
    const int w    = tid >> 5;               // wave id (split-K slice)
    const int lane = tid & 31;
    const int q    = lane & 15;              // query col / A row / channel col
    const int h    = lane >> 4;

    // B-layout Q^T: lane half h elems 0-7 hold d=8h..8h+7; elems 8-15 zero padding.
    v16h qb = {};
    {
        const int qoff = (b * N_ + row0 + q) * D_ + 8 * h;
        #pragma unroll
        for (int j = 0; j < 8; ++j) qb[j] = Qh[qoff + j];
    }

    float mrun = -INFINITY, lrun = 0.f;      // per-query (per-lane) scalars
    v8f   acc[8];                            // O^T: row=channel(r+8h), col=q
    const v8f vzero = {0.f, 0.f, 0.f, 0.f, 0.f, 0.f, 0.f, 0.f};
    #pragma unroll
    for (int g = 0; g < 8; ++g) acc[g] = vzero;

    // A-row m' -> key m' + (m'&8): S1 row r+8h = key 16h+r,
    // ka2 (+8 rows)             : S2 row r+8h = key 16h+8+r.
    const int krow  = q + (q & 8);
    const int kbase = (b * N_ + krow) * D_ + 8 * h;   // 32-bit element offset
    const int vbase = (b * C_ + q) * N_ + 16 * h;     // 32-bit element offset

    for (int kt = w; kt < N_ / 32; kt += NW) {
        const int n0 = kt * 32;

        v16h ka1 = {}, ka2 = {};
        {
            const int ko = kbase + n0 * D_;
            #pragma unroll
            for (int j = 0; j < 8; ++j) {
                ka1[j] = Kh[ko + j];
                ka2[j] = Kh[ko + 8 * D_ + j];
            }
        }
        v8f S1 = __builtin_amdgcn_wmma_f32_16x16x32_f16(
            false, ka1, false, qb, (short)0, vzero, false, false);
        v8f S2 = __builtin_amdgcn_wmma_f32_16x16x32_f16(
            false, ka2, false, qb, (short)0, vzero, false, false);

        // --- online softmax (log2 domain): in-lane + 1 cross-half ---------
        float tm = fmaxf(S1[0], S2[0]);
        #pragma unroll
        for (int r = 1; r < 8; ++r) tm = fmaxf(tm, fmaxf(S1[r], S2[r]));
        tm = fmaxf(tm, __shfl_xor(tm, 16, 32));

        const float mn = fmaxf(mrun, tm);
        const float al = __builtin_exp2f(mrun - mn);
        mrun = mn;

        float p1[8], p2[8], ps = 0.f;
        #pragma unroll
        for (int r = 0; r < 8; ++r) {
            p1[r] = __builtin_exp2f(S1[r] - mn);   // key 16h + r
            p2[r] = __builtin_exp2f(S2[r] - mn);   // key 16h + 8 + r
            ps += p1[r] + p2[r];
        }
        ps += __shfl_xor(ps, 16, 32);
        lrun = lrun * al + ps;

        if (__any(al != 1.0f)) {                   // skip when max unchanged
            #pragma unroll
            for (int g = 0; g < 8; ++g)
                #pragma unroll
                for (int r = 0; r < 8; ++r)
                    acc[g][r] *= al;
        }

        // P^T in B layout: lane (q,h) elem j = key 16h+j -- all lane-owned.
        v16h pb;
        #pragma unroll
        for (int j = 0; j < 8; ++j) {
            pb[j]     = (half_t)p1[j];
            pb[j + 8] = (half_t)p2[j];
        }

        // O^T(128x16) += V^T(128x32) @ P^T(32x16); uniform base + 32-bit
        // offsets so loads take the saddr+imm form (g*16*N fits IOFFSET).
        const int vo = vbase + n0;
        #pragma unroll
        for (int g = 0; g < 8; ++g) {
            v16h va;
            #pragma unroll
            for (int j = 0; j < 8; ++j) {
                va[j]     = Vt[vo + g * (16 * N_) + j];
                va[j + 8] = Vt[vo + g * (16 * N_) + 8 + j];
            }
            acc[g] = __builtin_amdgcn_wmma_f32_16x16x32_f16(
                false, va, false, pb, (short)0, acc[g], false, false);
        }
    }

    // --- split-K merge through LDS (exp2 domain) --------------------------
    if (h == 0) { Ml[w][q] = mrun; Ll[w][q] = lrun; }
    __syncthreads();

    float mstar = Ml[0][q];
    #pragma unroll
    for (int w2 = 1; w2 < NW; ++w2) mstar = fmaxf(mstar, Ml[w2][q]);
    float lstar = 0.f;
    #pragma unroll
    for (int w2 = 0; w2 < NW; ++w2)
        lstar += Ll[w2][q] * __builtin_exp2f(Ml[w2][q] - mstar);
    if (w == 0 && h == 0) Lst[q] = lstar;

    const float myscale = __builtin_exp2f(mrun - mstar);
    #pragma unroll
    for (int g = 0; g < 8; ++g)
        #pragma unroll
        for (int r = 0; r < 8; ++r)
            Osts[w][q][g * 16 + r + 8 * h] = acc[g][r] * myscale;
    __syncthreads();

    // --- coalesced writeback: 2048 outputs over 128 threads ---------------
    const float gm = gamma[0];
    const int c0 = (tid & 31) * 4;           // 4 consecutive channels
    const int i0 = (tid >> 5) * 4;           // 4 rows
    #pragma unroll
    for (int ii = 0; ii < 4; ++ii) {
        const int    i    = i0 + ii;
        const float  rinv = 1.0f / Lst[i];
        const size_t idx  = (size_t)(b * N_ + row0 + i) * C_ + c0;
        #pragma unroll
        for (int u = 0; u < 4; ++u) {
            const float o = Osts[0][i][c0 + u] + Osts[1][i][c0 + u] +
                            Osts[2][i][c0 + u] + Osts[3][i][c0 + u];
            out[idx + u] = fmaf(gm, o * rinv, x[idx + u]);
        }
    }
}

// ---------------------------------------------------------------------------
extern "C" void kernel_launch(void* const* d_in, const int* in_sizes, int n_in,
                              void* d_out, int out_size, void* d_ws,
                              size_t ws_size, hipStream_t stream)
{
    (void)in_sizes; (void)n_in; (void)out_size; (void)ws_size;
    const float* x     = (const float*)d_in[0];
    const float* wq    = (const float*)d_in[1];
    const float* bq    = (const float*)d_in[2];
    const float* wk    = (const float*)d_in[3];
    const float* bk    = (const float*)d_in[4];
    const float* wv    = (const float*)d_in[5];
    const float* bv    = (const float*)d_in[6];
    const float* gamma = (const float*)d_in[7];

    half_t* Qh = (half_t*)d_ws;                       //  B*N*16 f16  (512 KB)
    half_t* Kh = Qh + (size_t)B_ * N_ * D_;           //  B*N*16 f16  (512 KB)
    half_t* Vt = Kh + (size_t)B_ * N_ * D_;           //  B*C*N  f16  (4 MB)

    proj_kernel<<<B_ * N_ / 16, 128, 0, stream>>>(x, wq, bq, wk, bk, wv, bv,
                                                  Qh, Kh, Vt);
    attn_kernel<<<B_ * N_ / 16, 32 * NW, 0, stream>>>(Qh, Kh, Vt, x, gamma,
                                                      (float*)d_out);
}
